// MoELayer_71047349010620
// MI455X (gfx1250) — matrix operand order repaired and checked by reading, not compiled, not run on hardware
//
#include <hip/hip_runtime.h>
#include <hip/hip_bf16.h>

#define HD 1024
#define ID 2048
#define NE 8
#define NT 4096   // total tokens = 2*2048

typedef __attribute__((ext_vector_type(16))) __bf16        v16bf;
typedef __attribute__((ext_vector_type(8)))  float         v8f;
typedef __attribute__((ext_vector_type(8)))  unsigned int  v8u;

__device__ __forceinline__ unsigned short f2bf(float f) {
  unsigned int u = __float_as_uint(f);
  return (unsigned short)((u + 0x7FFFu + ((u >> 16) & 1u)) >> 16);
}
__device__ __forceinline__ unsigned int f2bf_pk(float lo, float hi) {
  return (unsigned int)f2bf(lo) | ((unsigned int)f2bf(hi) << 16);
}

// B-operand pair fetch: bf16 weights -> raw dword; f32 weights -> pack on the fly
__device__ __forceinline__ unsigned int pack2(const unsigned short* p) {
  return *(const unsigned int*)p;
}
__device__ __forceinline__ unsigned int pack2(const float* p) {
  return f2bf_pk(p[0], p[1]);
}

// ---------------------------------------------------------------------------
// fp32 -> bf16 bulk conversion (one-shot per launch; bandwidth bound)
// ---------------------------------------------------------------------------
__global__ void __launch_bounds__(256)
cvt_bf16_kernel(const float* __restrict__ src, unsigned short* __restrict__ dst,
                int n4)
{
  int i = blockIdx.x * blockDim.x + threadIdx.x;
  const int stride = gridDim.x * blockDim.x;
  for (; i < n4; i += stride) {
    const float4 f = ((const float4*)src)[i];
    uint2 o;
    o.x = f2bf_pk(f.x, f.y);
    o.y = f2bf_pk(f.z, f.w);
    ((uint2*)dst)[i] = o;
  }
}

// ---------------------------------------------------------------------------
// Router: layernorm -> logits -> softmax -> top2 -> combine weights w[T,E].
// Also emits bf16 copies of raw x (routed experts) and sh-layernormed x
// (shared expert), and accumulates sum(z^2) for the z-loss.
// One wave32 per token; 8 waves per block.
// ---------------------------------------------------------------------------
__global__ void __launch_bounds__(256)
router_kernel(const float* __restrict__ x,
              const float* __restrict__ ln_g, const float* __restrict__ ln_b,
              const float* __restrict__ gate_w, const float* __restrict__ ebias,
              const float* __restrict__ sh_g, const float* __restrict__ sh_b,
              float* __restrict__ w_buf,
              unsigned short* __restrict__ xb,
              unsigned short* __restrict__ xsb,
              float* __restrict__ zacc)
{
  const int lane = threadIdx.x & 31;
  const int wv   = threadIdx.x >> 5;
  const int t    = blockIdx.x * 8 + wv;
  if (t >= NT) return;

  const float* xt = x + (size_t)t * HD;
  float xr[32];
  #pragma unroll
  for (int i = 0; i < 32; ++i) xr[i] = xt[lane + 32 * i];

  float s = 0.f;
  #pragma unroll
  for (int i = 0; i < 32; ++i) s += xr[i];
  #pragma unroll
  for (int off = 16; off >= 1; off >>= 1) s += __shfl_xor(s, off, 32);
  const float mean = s * (1.0f / HD);

  float v = 0.f;
  #pragma unroll
  for (int i = 0; i < 32; ++i) { float d = xr[i] - mean; v += d * d; }
  #pragma unroll
  for (int off = 16; off >= 1; off >>= 1) v += __shfl_xor(v, off, 32);
  const float rstd = rsqrtf(v * (1.0f / HD) + 1e-5f);

  float hn[32];
  #pragma unroll
  for (int i = 0; i < 32; ++i) {
    int h = lane + 32 * i;
    hn[i] = (xr[i] - mean) * rstd * ln_g[h] + ln_b[h];
  }

  float logits[NE];
  #pragma unroll
  for (int e = 0; e < NE; ++e) {
    float acc = 0.f;
    #pragma unroll
    for (int i = 0; i < 32; ++i) acc += hn[i] * gate_w[e * HD + lane + 32 * i];
    #pragma unroll
    for (int off = 16; off >= 1; off >>= 1) acc += __shfl_xor(acc, off, 32);
    logits[e] = acc + ebias[e];
  }

  float mx = logits[0];
  #pragma unroll
  for (int e = 1; e < NE; ++e) mx = fmaxf(mx, logits[e]);
  float p[NE], se = 0.f;
  #pragma unroll
  for (int e = 0; e < NE; ++e) { p[e] = __expf(logits[e] - mx); se += p[e]; }
  int i0 = 0; float p0 = p[0];
  #pragma unroll
  for (int e = 1; e < NE; ++e) if (p[e] > p0) { p0 = p[e]; i0 = e; }
  int i1 = -1; float p1 = -1.f;
  #pragma unroll
  for (int e = 0; e < NE; ++e) if (e != i0 && p[e] > p1) { p1 = p[e]; i1 = e; }
  const float inv = 1.0f / se;
  p0 *= inv; p1 *= inv;
  float denom = p0 + p1; denom = (denom < 1e-5f) ? 1e-5f : denom;
  const float w0 = p0 / denom, w1 = p1 / denom;

  if (lane < NE) {
    float wvw = (lane == i0) ? w0 : ((lane == i1) ? w1 : 0.f);
    w_buf[(size_t)t * NE + lane] = wvw;
  }
  if (lane == 0) {
    float z = mx + __logf(se);
    atomicAdd(zacc, z * z);
  }

  #pragma unroll
  for (int i = 0; i < 32; ++i) {
    int h = lane + 32 * i;
    xb[(size_t)t * HD + h] = f2bf(xr[i]);
    float xs = (xr[i] - mean) * rstd * sh_g[h] + sh_b[h];
    xsb[(size_t)t * HD + h] = f2bf(xs);
  }
}

// ---------------------------------------------------------------------------
// Fused MoE experts + shared expert. One WG = 16 tokens, 8 waves.
// e in [0..7]: routed expert (skipped if all 16 combine weights are zero),
// e == 8    : shared expert (scale = sigmoid(shared_gate)).
// Combine weight folded into the activation before the down projection so a
// single persistent f32 accumulator sums every expert. WT = unsigned short
// (pre-converted bf16, fast path) or float (pack-on-the-fly fallback).
// ---------------------------------------------------------------------------
template <typename WT>
__global__ void __launch_bounds__(256)
moe_expert_kernel(const unsigned short* __restrict__ xb,
                  const unsigned short* __restrict__ xsb,
                  const WT* __restrict__ Wg, const WT* __restrict__ Wu,
                  const WT* __restrict__ Wd,
                  const WT* __restrict__ shWg, const WT* __restrict__ shWu,
                  const WT* __restrict__ shWd,
                  const float* __restrict__ shared_gate,
                  const float* __restrict__ w_buf,
                  float* __restrict__ out)
{
  __shared__ unsigned short sAct[16 * 260];   // 16 x 256 bf16, +4 pad/row

  const int lane = threadIdx.x & 31;
  const int wv   = threadIdx.x >> 5;   // 0..7
  const int hi   = lane >> 4;          // 0/1 (lane half)
  const int ln   = lane & 15;
  const int t0   = blockIdx.x * 16;

  const v8f zero = {};
  v8f oacc[8];
  #pragma unroll
  for (int n = 0; n < 8; ++n) oacc[n] = zero;

  const float sgate = 1.0f / (1.0f + __expf(-shared_gate[0]));

  for (int e = 0; e <= NE; ++e) {
    float rowW[8];
    const unsigned short* A0;
    const WT *Pg, *Pu, *Pd;
    if (e < NE) {
      float wt = (lane < 16) ? w_buf[(size_t)(t0 + lane) * NE + e] : 0.f;
      if (!__any(wt != 0.f)) continue;            // uniform across the WG
      #pragma unroll
      for (int r = 0; r < 8; ++r)
        rowW[r] = w_buf[(size_t)(t0 + r + 8 * hi) * NE + e];
      A0 = xb;
      Pg = Wg + (size_t)e * ID * HD;
      Pu = Wu + (size_t)e * ID * HD;
      Pd = Wd + (size_t)e * HD * ID;
    } else {
      #pragma unroll
      for (int r = 0; r < 8; ++r) rowW[r] = sgate;
      A0 = xsb; Pg = shWg; Pu = shWu; Pd = shWd;
    }

    for (int ci = 0; ci < 8; ++ci) {              // 8 chunks of 256 I-cols
      const int cibase = ci * 256;

      // ---- phase 1: gate/up GEMM for this wave's 32 I-columns ----
      v8f ga[2] = { zero, zero };
      v8f ua[2] = { zero, zero };
      for (int kh = 0; kh < HD; kh += 32) {
        v8u ar;
        #pragma unroll
        for (int j = 0; j < 8; ++j) {
          int k0 = 16 * hi + 2 * j;
          ar[j] = *(const unsigned int*)(A0 + (size_t)(t0 + ln) * HD + kh + k0);
        }
        v16bf a = __builtin_bit_cast(v16bf, ar);
        #pragma unroll
        for (int nt = 0; nt < 2; ++nt) {
          const int n = cibase + wv * 32 + nt * 16 + ln;
          v8u bg, bu;
          #pragma unroll
          for (int j = 0; j < 8; ++j) {
            int k0 = kh + 16 * hi + 2 * j;
            bg[j] = pack2(Pg + (size_t)n * HD + k0);
            bu[j] = pack2(Pu + (size_t)n * HD + k0);
          }
          ga[nt] = __builtin_amdgcn_wmma_f32_16x16x32_bf16(
              false, a, false, __builtin_bit_cast(v16bf, bg),
              (short)0, ga[nt], false, false);
          ua[nt] = __builtin_amdgcn_wmma_f32_16x16x32_bf16(
              false, a, false, __builtin_bit_cast(v16bf, bu),
              (short)0, ua[nt], false, false);
        }
      }

      // silu(g)*u, pre-scaled by combine weight -> LDS (bf16)
      #pragma unroll
      for (int nt = 0; nt < 2; ++nt) {
        #pragma unroll
        for (int r = 0; r < 8; ++r) {
          float g = ga[nt][r], u = ua[nt][r];
          float act = g * (1.0f / (1.0f + __expf(-g))) * u * rowW[r];
          int M = r + 8 * hi;
          int col = wv * 32 + nt * 16 + ln;
          sAct[M * 260 + col] = f2bf(act);
        }
      }
      __syncthreads();

      // ---- phase 2: down-proj, this wave owns 128 H-columns ----
      for (int kk = 0; kk < 256; kk += 32) {
        v8u ar;
        #pragma unroll
        for (int j = 0; j < 8; ++j) {
          int k0 = 16 * hi + 2 * j + kk;
          ar[j] = *(const unsigned int*)&sAct[ln * 260 + k0];
        }
        v16bf a = __builtin_bit_cast(v16bf, ar);
        #pragma unroll
        for (int nt = 0; nt < 8; ++nt) {
          const int n = wv * 128 + nt * 16 + ln;     // output H column
          __builtin_prefetch(Pd + (size_t)n * ID + cibase + kk + 32, 0, 1);
          v8u bd;
          #pragma unroll
          for (int j = 0; j < 8; ++j) {
            int k0 = cibase + kk + 16 * hi + 2 * j;
            bd[j] = pack2(Pd + (size_t)n * ID + k0);
          }
          oacc[nt] = __builtin_amdgcn_wmma_f32_16x16x32_bf16(
              false, a, false, __builtin_bit_cast(v16bf, bd),
              (short)0, oacc[nt], false, false);
        }
      }
      __syncthreads();
    }
  }

  // final store: routed + shared already summed in oacc
  #pragma unroll
  for (int nt = 0; nt < 8; ++nt) {
    const int n = wv * 128 + nt * 16 + ln;
    #pragma unroll
    for (int r = 0; r < 8; ++r) {
      int tok = t0 + r + 8 * hi;
      out[(size_t)tok * HD + n] = oacc[nt][r];
    }
  }
}

__global__ void zinit_kernel(float* zacc) {
  if (threadIdx.x == 0 && blockIdx.x == 0) zacc[0] = 0.f;
}
__global__ void aux_kernel(const float* __restrict__ zacc, float* __restrict__ aux) {
  if (threadIdx.x == 0 && blockIdx.x == 0)
    aux[0] = zacc[0] * (1e-4f / (float)NT);
}

extern "C" void kernel_launch(void* const* d_in, const int* in_sizes, int n_in,
                              void* d_out, int out_size, void* d_ws, size_t ws_size,
                              hipStream_t stream)
{
  const float* x        = (const float*)d_in[0];
  const float* ln_g     = (const float*)d_in[1];
  const float* ln_b     = (const float*)d_in[2];
  const float* gate_w   = (const float*)d_in[3];
  const float* ebias    = (const float*)d_in[4];
  const float* Wg       = (const float*)d_in[5];
  const float* Wu       = (const float*)d_in[6];
  const float* Wd       = (const float*)d_in[7];
  const float* sh_ln_g  = (const float*)d_in[8];
  const float* sh_ln_b  = (const float*)d_in[9];
  const float* shWg     = (const float*)d_in[10];
  const float* shWu     = (const float*)d_in[11];
  const float* shWd     = (const float*)d_in[12];
  const float* sh_gate  = (const float*)d_in[13];

  char* ws = (char*)d_ws;
  size_t off = 0;
  float* w_buf = (float*)(ws + off);  off += (size_t)NT * NE * sizeof(float);
  float* zacc  = (float*)(ws + off);  off += 256;
  unsigned short* xb  = (unsigned short*)(ws + off); off += (size_t)NT * HD * 2;
  unsigned short* xsb = (unsigned short*)(ws + off); off += (size_t)NT * HD * 2;

  const size_t wcnt  = (size_t)NE * ID * HD;   // elems per routed weight tensor
  const size_t scnt  = (size_t)ID * HD;        // elems per shared weight tensor
  unsigned short* bWg  = (unsigned short*)(ws + off);
  unsigned short* bWu  = bWg  + wcnt;
  unsigned short* bWd  = bWu  + wcnt;
  unsigned short* bsWg = bWd  + wcnt;
  unsigned short* bsWu = bsWg + scnt;
  unsigned short* bsWd = bsWu + scnt;
  const size_t need = off + (3 * wcnt + 3 * scnt) * 2;

  float* out_f = (float*)d_out;
  float* aux   = out_f + (size_t)NT * HD;

  zinit_kernel<<<1, 32, 0, stream>>>(zacc);
  router_kernel<<<NT / 8, 256, 0, stream>>>(x, ln_g, ln_b, gate_w, ebias,
                                            sh_ln_g, sh_ln_b,
                                            w_buf, xb, xsb, zacc);

  if (ws_size >= need) {
    // fast path: mirror all weights to bf16 once (then L2-resident: 113MB < 192MB)
    const int CB = 2048, CT = 256;
    cvt_bf16_kernel<<<CB, CT, 0, stream>>>(Wg,   bWg,  (int)(wcnt / 4));
    cvt_bf16_kernel<<<CB, CT, 0, stream>>>(Wu,   bWu,  (int)(wcnt / 4));
    cvt_bf16_kernel<<<CB, CT, 0, stream>>>(Wd,   bWd,  (int)(wcnt / 4));
    cvt_bf16_kernel<<<CB, CT, 0, stream>>>(shWg, bsWg, (int)(scnt / 4));
    cvt_bf16_kernel<<<CB, CT, 0, stream>>>(shWu, bsWu, (int)(scnt / 4));
    cvt_bf16_kernel<<<CB, CT, 0, stream>>>(shWd, bsWd, (int)(scnt / 4));
    moe_expert_kernel<unsigned short><<<NT / 16, 256, 0, stream>>>(
        xb, xsb, bWg, bWu, bWd, bsWg, bsWu, bsWd, sh_gate, w_buf, out_f);
  } else {
    // fallback: pack fp32 weights to bf16 inside the GEMM
    moe_expert_kernel<float><<<NT / 16, 256, 0, stream>>>(
        xb, xsb, Wg, Wu, Wd, shWg, shWu, shWd, sh_gate, w_buf, out_f);
  }

  aux_kernel<<<1, 32, 0, stream>>>(zacc, aux);
}
